// RESCAL_60868276518931
// MI455X (gfx1250) — matrix operand order repaired and checked by reading, not compiled, not run on hardware
//
#include <hip/hip_runtime.h>

// ---------------------------------------------------------------------------
// Problem sizes (fixed by the reference)
// ---------------------------------------------------------------------------
#define N_ENT   2048
#define N_REL   32
#define DIM     512
#define BTILE   128     // output tile per block (128 x 128), 8 waves
#define KSTEP   32      // K per WMMA
#define APITCH  40      // LDS row pitch in bf16 (80B = 5*16B: 16B-aligned b128
                        // fragment loads, conflict-free 20-dword row stride)
#define NKS     (DIM / KSTEP)   // 16 K-steps

typedef __attribute__((ext_vector_type(16))) __bf16 v16bf;
typedef __attribute__((ext_vector_type(8)))  __bf16 v8bf;
typedef __attribute__((ext_vector_type(8)))  float  v8f;
typedef __attribute__((ext_vector_type(4)))  unsigned int v4u;
typedef __attribute__((ext_vector_type(8)))  int v8i;
typedef __attribute__((ext_vector_type(4)))  int v4i;

// ---------------------------------------------------------------------------
// Kernel 1: E = X @ W^T + b for X = [entities ; relations]  (0.8% of FLOPs,
// plain f32 FMA). Writes f32 E/R plus a bf16 copy of E for the WMMA einsum.
// ---------------------------------------------------------------------------
__global__ __launch_bounds__(256) void fused_linear(
    const float* __restrict__ ent, const float* __restrict__ rel,
    const float* __restrict__ W,   const float* __restrict__ b,
    float* __restrict__ Ef32, float* __restrict__ Rf32,
    __bf16* __restrict__ Ebf)
{
    __shared__ float xrow[DIM];
    const int i = blockIdx.x;                    // 0..2079
    const int t = threadIdx.x;                   // 0..255
    const float* src = (i < N_ENT) ? (ent + (size_t)i * DIM)
                                   : (rel + (size_t)(i - N_ENT) * DIM);
    for (int d = t; d < DIM; d += 256) xrow[d] = src[d];
    __syncthreads();

    for (int j = t; j < DIM; j += 256) {
        const float4* Wr = (const float4*)(W + (size_t)j * DIM);
        float acc = 0.f;
#pragma unroll 8
        for (int q = 0; q < DIM / 4; ++q) {
            const float4 w4 = Wr[q];
            const float4 x4 = *(const float4*)(&xrow[q * 4]);
            acc += w4.x * x4.x + w4.y * x4.y + w4.z * x4.z + w4.w * x4.w;
        }
        const float val = acc + b[j];
        if (i < N_ENT) {
            Ef32[(size_t)i * DIM + j] = val;
            Ebf [(size_t)i * DIM + j] = (__bf16)val;
        } else {
            Rf32[(size_t)(i - N_ENT) * DIM + j] = val;
        }
    }
}

// ---------------------------------------------------------------------------
// Kernel 2: scores[e,r,f] = sum_d (E[e,d]*R[r,d]) * E[f,d]
//   per relation r:  S_r = A_r @ E^T,  A_r = E * R_r (row-broadcast scale)
// 256 threads = 8 waves; 128x128 tile; wave = 32(m) x 64(n) = 2x4 WMMAs.
// Double-buffered LDS, one barrier per K-step.
//   A tile: VGPR path (needs scale by R_r + f32->bf16 cvt)
//   B tile: TDM tensor_load_to_lds (TENSORcnt), descriptor pads LDS rows
//           64B data + 16B pad = 80B pitch; prologue uses per-lane
//           global_load_async_to_lds_b128 (ASYNCcnt).
// ---------------------------------------------------------------------------
__global__ __launch_bounds__(256) void rescal_main(
    const float* __restrict__ Ef32, const float* __restrict__ Rf32,
    const __bf16* __restrict__ Ebf, float* __restrict__ out)
{
    __shared__ __bf16 As[2][BTILE][APITCH];   // (E*R_r) rows e, cols k
    __shared__ __bf16 Bs[2][BTILE][APITCH];   // rows f, cols k (= E^T col-major)
    __shared__ float  Rrow[DIM];

    const int tf = blockIdx.x * BTILE;        // f-tile origin
    const int te = blockIdx.y * BTILE;        // e-tile origin
    const int r  = blockIdx.z;                // relation
    const int t    = threadIdx.x;             // 0..255
    const int wave = t >> 5;
    const int lane = t & 31;
    const int half = lane >> 4;
    const int l15  = lane & 15;
    const int m0 = (wave >> 1) * 32;          // 4 waves along m
    const int n0 = (wave & 1)  * 64;          // 2 waves along n

    for (int d = t; d < DIM; d += 256) Rrow[d] = Rf32[(size_t)r * DIM + d];
    __syncthreads();

    // per-thread staging coordinates
    const int arow = t >> 3;                  // A: +32 rows per iter (4 iters)
    const int ac4  = (t & 7) << 2;            // A: float4 column
    const int brow = t >> 2;                  // B: +64 rows per iter (2 iters)
    const int bc8  = (t & 3) << 3;            // B: 8-bf16 column

    // ---- staging helpers -------------------------------------------------
    auto stageA_store = [&](const float4* v, int buf, int k0) {
#pragma unroll
        for (int it = 0; it < 4; ++it) {
            const int row = arow + it * 32;
            As[buf][row][ac4 + 0] = (__bf16)(v[it].x * Rrow[k0 + ac4 + 0]);
            As[buf][row][ac4 + 1] = (__bf16)(v[it].y * Rrow[k0 + ac4 + 1]);
            As[buf][row][ac4 + 2] = (__bf16)(v[it].z * Rrow[k0 + ac4 + 2]);
            As[buf][row][ac4 + 3] = (__bf16)(v[it].w * Rrow[k0 + ac4 + 3]);
        }
    };
    auto loadA_regs = [&](float4* v, int k0) {
#pragma unroll
        for (int it = 0; it < 4; ++it)
            v[it] = *(const float4*)(&Ef32[(size_t)(te + arow + it * 32) * DIM + k0 + ac4]);
    };
    auto asyncB = [&](int buf, int k0) {      // per-lane ASYNCcnt path (prologue)
#pragma unroll
        for (int it = 0; it < 2; ++it) {
            const int row = brow + it * 64;
            const unsigned long long ga =
                (unsigned long long)(&Ebf[(size_t)(tf + row) * DIM + k0 + bc8]);
            const unsigned la = (unsigned)(size_t)(&Bs[buf][row][bc8]);
            asm volatile("global_load_async_to_lds_b128 %0, %1, off"
                         :: "v"(la), "v"(ga) : "memory");
        }
    };
    // TDM: one DMA moves the whole 128x32 bf16 B tile; D# pad feature writes
    // each 64B row then skips 16B -> 80B LDS pitch (APITCH).
    auto tdmB = [&](int buf, int k0) {
        const unsigned long long ga =
            (unsigned long long)(size_t)(&Ebf[(size_t)tf * DIM + k0]);
        const unsigned la = (unsigned)(size_t)(&Bs[buf][0][0]);
        v4u g0;
        g0[0] = 1u;                                   // count=1, user mode
        g0[1] = la;                                   // lds_addr
        g0[2] = (unsigned)(ga & 0xFFFFFFFFu);         // global_addr[31:0]
        g0[3] = (unsigned)((ga >> 32) & 0x1FFFFFFu)   // global_addr[56:32]
              | (2u << 30);                           // type = 2 ("image")
        v8i g1;
        g1[0] = (1 << 16)      // data_size = 2 bytes
              | (1 << 20)      // pad_enable
              | (3 << 22)      // pad_interval: every 16 DWORDs (64B)
              | (3 << 25);     // pad_amount: 4 DWORDs (16B)
        g1[1] = (DIM & 0xFFFF) << 16;                 // tensor_dim0 = 512 (lo16)
        g1[2] = (N_ENT & 0xFFFF) << 16;               // dim0 hi=0 | tensor_dim1 lo16
        g1[3] = (KSTEP << 16);                        // dim1 hi=0 | tile_dim0 = 32
        g1[4] = BTILE;                                // tile_dim1 = 128, tile_dim2 = 0
        g1[5] = DIM;                                  // tensor_dim0_stride = 512
        g1[6] = 0;                                    // stride0 hi | stride1 lo
        g1[7] = 0;                                    // stride1 hi
        const v4i gz4 = {0, 0, 0, 0};                 // 2D tensor: groups 2/3 unused
        const v8i gz8 = {0, 0, 0, 0, 0, 0, 0, 0};
        __builtin_amdgcn_tensor_load_to_lds(g0, g1, gz4, gz4, gz8, 0);
    };

    // ---- prologue: stage tile 0 into buffer 0 ---------------------------
    {
        float4 a0[4];
        loadA_regs(a0, 0);
        asyncB(0, 0);
        stageA_store(a0, 0, 0);
    }
    asm volatile("s_wait_asynccnt 0x0" ::: "memory");
    __syncthreads();

    const v8f vzero = {0.f, 0.f, 0.f, 0.f, 0.f, 0.f, 0.f, 0.f};
    v8f acc[2][4] = {{vzero, vzero, vzero, vzero}, {vzero, vzero, vzero, vzero}};

    for (int ks = 0; ks < NKS; ++ks) {
        const int cur = ks & 1;
        const int nxt = cur ^ 1;
        const int k1  = (ks + 1) * KSTEP;
        const bool have_next = (ks + 1 < NKS);

        // issue next tile's loads before computing (pipeline)
        float4 areg[4];
        if (have_next) {
            loadA_regs(areg, k1);
            if (wave == 0) tdmB(nxt, k1);     // TDM issues once per block
        }

        // ---- fragment loads (ISA 7.12.2 16-bit layouts) ----
        // A 16x32: half=0 -> k 0..7 & 16..23 ; half=1 -> k 8..15 & 24..31
        v16bf afrag[2], bfrag[4];
#pragma unroll
        for (int mt = 0; mt < 2; ++mt) {
            const int mrow = m0 + mt * 16 + l15;
            const v8bf lo = *(const v8bf*)(&As[cur][mrow][half * 8]);
            const v8bf hi = *(const v8bf*)(&As[cur][mrow][16 + half * 8]);
            afrag[mt] = __builtin_shufflevector(lo, hi, 0, 1, 2, 3, 4, 5, 6, 7,
                                                8, 9, 10, 11, 12, 13, 14, 15);
        }
        // B 32x16: lane = column n; half=0 -> k 0..7 & 16..23 ; half=1 -> k 8..15 & 24..31
#pragma unroll
        for (int nt = 0; nt < 4; ++nt) {
            const int ncol = n0 + nt * 16 + l15;
            const v8bf lo = *(const v8bf*)(&Bs[cur][ncol][half * 8]);
            const v8bf hi = *(const v8bf*)(&Bs[cur][ncol][16 + half * 8]);
            bfrag[nt] = __builtin_shufflevector(lo, hi, 0, 1, 2, 3, 4, 5, 6, 7,
                                                8, 9, 10, 11, 12, 13, 14, 15);
        }

        // ---- 8x v_wmma_f32_16x16x32_bf16 ----
#pragma unroll
        for (int mt = 0; mt < 2; ++mt)
#pragma unroll
            for (int nt = 0; nt < 4; ++nt)
                acc[mt][nt] = __builtin_amdgcn_wmma_f32_16x16x32_bf16(
                    false, afrag[mt], false, bfrag[nt],
                    (short)0, acc[mt][nt], false, false);

        // convert + store next A tile into the other buffer
        if (have_next)
            stageA_store(areg, nxt, k1);

        __builtin_amdgcn_s_wait_tensorcnt(0);   // no-op for waves 1..7
        __syncthreads();
    }

    // ---- store: C/D layout: VGPR i, lanes 0-15 -> M=i, lanes 16-31 -> M=8+i
    // scores layout [e][r][f]; streamed write-once -> non-temporal
#pragma unroll
    for (int mt = 0; mt < 2; ++mt) {
#pragma unroll
        for (int nt = 0; nt < 4; ++nt) {
            const int f = tf + n0 + nt * 16 + l15;
#pragma unroll
            for (int i = 0; i < 8; ++i) {
                const int e = te + m0 + mt * 16 + i + 8 * half;
                const size_t idx = ((size_t)e * N_REL + r) * (size_t)N_ENT + f;
                __builtin_nontemporal_store(acc[mt][nt][i], &out[idx]);
            }
        }
    }
}

// ---------------------------------------------------------------------------
// Launch
// ---------------------------------------------------------------------------
extern "C" void kernel_launch(void* const* d_in, const int* in_sizes, int n_in,
                              void* d_out, int out_size, void* d_ws, size_t ws_size,
                              hipStream_t stream)
{
    const float* entities  = (const float*)d_in[0];   // [2048,512]
    const float* relations = (const float*)d_in[1];   // [32,512]
    const float* W         = (const float*)d_in[2];   // [512,512]
    const float* b         = (const float*)d_in[3];   // [512]
    float* out = (float*)d_out;                       // [2048,32,2048]

    // workspace layout: Ef32 (4MB) | Rf32 (64KB) | Ebf16 (2MB)
    float*  Ef32 = (float*)d_ws;
    float*  Rf32 = Ef32 + (size_t)N_ENT * DIM;
    __bf16* Ebf  = (__bf16*)(Rf32 + (size_t)N_REL * DIM);

    fused_linear<<<N_ENT + N_REL, 256, 0, stream>>>(entities, relations, W, b,
                                                    Ef32, Rf32, Ebf);

    dim3 grid(N_ENT / BTILE, N_ENT / BTILE, N_REL);   // (16, 16, 32)
    rescal_main<<<grid, 256, 0, stream>>>(Ef32, Rf32, Ebf, out);
}
